// DecoderRNN_60533269070110
// MI455X (gfx1250) — matrix operand order repaired and checked by reading, not compiled, not run on hardware
//
#include <hip/hip_runtime.h>

// ---------------------------------------------------------------------------
// DecoderRNN (Bahdanau attention + LSTM + vocab projection) for MI455X/gfx1250
// - all matmuls via v_wmma_f32_16x16x32_bf16 (bf16 in, fp32 accumulate)
// - each wave computes a 64x32 macro-tile (4x2 WMMA tiles) for register reuse
// - B slab streamed into LDS with the Tensor Data Mover (tensor_load_to_lds),
//   double-buffered across 256-col K-chunks, s_wait_tensorcnt pipelining,
//   B fragments read back on the DS pipe (ds_load_b128)
// ---------------------------------------------------------------------------

enum { Bb = 64, Tt = 32, Ll = 49, Ff = 2048, Ee = 512, Hh = 512, Vv = 32000 };

typedef __attribute__((ext_vector_type(16))) __bf16 v16bf;
typedef __attribute__((ext_vector_type(8)))  float  v8f;
typedef __attribute__((ext_vector_type(4)))  unsigned int u32x4;
typedef __attribute__((ext_vector_type(8)))  int i32x8;
typedef __attribute__((ext_vector_type(4)))  int i32x4;

// LDS (address_space(3)) typed pointers for the B-fragment reads
typedef __attribute__((address_space(3))) const unsigned short lds_cus;
typedef __attribute__((address_space(3))) const unsigned int   lds_cui;

union FragU {
    unsigned int u[8];
    v16bf        b;
};

// GEMM macro-tile configuration
#define GMT 4            // M tiles per wave  (64 rows)
#define GNT 2            // N tiles per wave  (32 cols)
#define GKC 256          // K chunk staged in LDS (all K are multiples of 256)

#if defined(__HIP_DEVICE_COMPILE__) && \
    __has_builtin(__builtin_amdgcn_tensor_load_to_lds) && \
    __has_builtin(__builtin_amdgcn_s_wait_tensorcnt)
#define USE_TDM 1
#else
#define USE_TDM 0
#endif

__device__ __forceinline__ unsigned short f32_to_bf16(float f) {
    unsigned int u = __builtin_bit_cast(unsigned int, f);
    unsigned int r = u + 0x7FFFu + ((u >> 16) & 1u);   // round-to-nearest-even
    return (unsigned short)(r >> 16);
}

__device__ __forceinline__ float sigmoidf_(float x) {
    return 1.0f / (1.0f + __expf(-x));
}

// ------------------------- generic fp32 -> bf16 convert ---------------------
__global__ void cvt_f32_bf16_kernel(const float* __restrict__ in,
                                    unsigned short* __restrict__ out, int n) {
    int i = blockIdx.x * blockDim.x + threadIdx.x;
    if (i < n) out[i] = f32_to_bf16(in[i]);
}

// ------------------------- embedding gather (to bf16) -----------------------
__global__ void gather_embed_kernel(const float* __restrict__ emb_table,
                                    const int* __restrict__ captions,
                                    unsigned short* __restrict__ embed_bf) {
    int i = blockIdx.x * blockDim.x + threadIdx.x;       // over B*T*E
    if (i >= Bb * Tt * Ee) return;
    int e  = i % Ee;
    int bt = i / Ee;
    int tok = captions[bt];
    embed_bf[i] = f32_to_bf16(emb_table[(size_t)tok * Ee + e]);
}

// ------------------------- mean over L (to bf16) ----------------------------
__global__ void mean_features_kernel(const float* __restrict__ features,
                                     unsigned short* __restrict__ mean_bf) {
    int i = blockIdx.x * blockDim.x + threadIdx.x;       // over B*F
    if (i >= Bb * Ff) return;
    int f = i % Ff;
    int b = i / Ff;
    float s = 0.f;
    for (int l = 0; l < Ll; ++l)
        s += features[((size_t)b * Ll + l) * Ff + f];
    mean_bf[i] = f32_to_bf16(s * (1.0f / (float)Ll));
}

#if USE_TDM
// ---------------------------------------------------------------------------
// Issue a TDM 2D tile load: rows x cols bf16 elements, row stride = row_stride
// elements in global memory, packed row-major (row stride = cols) in LDS at
// byte offset lds_off. D# layout per CDNA5 ISA 8.3-8.6.
// clang-23 form: (u32x4 g0, i32x8 g1, i32x4 g2, i32x4 g3, i32x8 g4, i32 cpol)
// ---------------------------------------------------------------------------
__device__ __forceinline__ void tdm_load_2d(const void* gsrc, unsigned lds_off,
                                            int rows, int cols, int row_stride) {
    unsigned long long ga = (unsigned long long)gsrc;
    u32x4 g0;
    g0[0] = 1u;                                            // count=1 (valid user D#)
    g0[1] = lds_off;                                       // lds_addr (bytes)
    g0[2] = (unsigned)ga;                                  // global_addr[31:0]
    g0[3] = (unsigned)((ga >> 32) & 0x01FFFFFFull)         // global_addr[56:32]
          | (2u << 30);                                    // type=2 ("image")
    i32x8 g1;
    g1[0] = (int)(1u << 16);                               // data_size=1 -> 2 bytes
    g1[1] = (int)(((unsigned)cols & 0xFFFFu) << 16);       // tensor_dim0[15:0]
    g1[2] = (int)((((unsigned)cols >> 16) & 0xFFFFu)       // tensor_dim0[31:16]
          | (((unsigned)rows & 0xFFFFu) << 16));           // tensor_dim1[15:0]
    g1[3] = (int)((((unsigned)rows >> 16) & 0xFFFFu)       // tensor_dim1[31:16]
          | (((unsigned)cols & 0xFFFFu) << 16));           // tile_dim0
    g1[4] = (int)((unsigned)rows & 0xFFFFu);               // tile_dim1 (tile_dim2=0)
    g1[5] = (int)(unsigned)row_stride;                     // tensor_dim0_stride[31:0]
    g1[6] = 0;                                             // stride[47:32]=0, dim1_stride lo
    g1[7] = 0;
    i32x4 gz4 = {0, 0, 0, 0};
    i32x8 gz8 = {0, 0, 0, 0, 0, 0, 0, 0};
    __builtin_amdgcn_tensor_load_to_lds(g0, g1, gz4, gz4, gz8, 0);
}
#endif

__device__ __forceinline__ void copy_slab_sync(const unsigned short* __restrict__ g,
                                               unsigned short* __restrict__ s,
                                               int ldb, int lane) {
    for (int r = 0; r < GNT * 16; ++r) {
        const unsigned int* gr = (const unsigned int*)(g + (size_t)r * ldb);
        unsigned int* sr = (unsigned int*)(s + r * GKC);
#pragma unroll
        for (int c = 0; c < (GKC / 2) / 32; ++c)
            sr[lane + 32 * c] = gr[lane + 32 * c];
    }
}

// ---------------------------------------------------------------------------
// WMMA GEMM:  C[M,N] (fp32) = A[M,K](bf16,row-major,lda) @ B[N,K](bf16)^T
//             + bias0[n] + bias1[n] (+ C if accumulate)
// One wave per block; each wave owns a 64x32 output macro-tile (4x2 WMMA
// tiles).  B slab (32 x GKC bf16) staged in LDS via TDM, double-buffered.
// Requires: M % 64 == 0, N % 32 == 0, K % GKC == 0 (true at all call sites).
// ---------------------------------------------------------------------------
__global__ void __launch_bounds__(32)
gemm_bf16_wmma_kernel(const unsigned short* __restrict__ A, int lda,
                      const unsigned short* __restrict__ B, int ldb,
                      float* __restrict__ C, long long ldc,
                      const float* __restrict__ bias0,
                      const float* __restrict__ bias1,
                      int K, int accumulate) {
    // double-buffered B slab: [2][32 rows][GKC cols] bf16  (32 KB)
    __shared__ unsigned short bsm[2][GNT * 16 * GKC];

    const int lane  = threadIdx.x & 31;
    const int half  = lane >> 4;     // 0: lanes 0-15, 1: lanes 16-31
    const int l16   = lane & 15;
    const int tileM = blockIdx.y * (16 * GMT);
    const int tileN = blockIdx.x * (16 * GNT);

    const int nchunks = K / GKC;

    v8f acc[GMT][GNT];
#pragma unroll
    for (int mt = 0; mt < GMT; ++mt)
#pragma unroll
        for (int nt = 0; nt < GNT; ++nt)
            acc[mt][nt] = (v8f){};

#if USE_TDM
    // prologue: kick off chunk 0  (bsm assumed at LDS offset 0: sole __shared__)
    tdm_load_2d(B + (size_t)tileN * ldb, 0u, GNT * 16, GKC, ldb);
#endif

    for (int ch = 0; ch < nchunks; ++ch) {
        const int k0 = ch * GKC;
        // LDS-typed pointer (AS3) so the fragment reads stay on the DS pipe
        lds_cus* bs = (lds_cus*)(bsm[ch & 1]);
#if USE_TDM
        if (ch + 1 < nchunks) {
            const unsigned lds_off = (unsigned)(((ch + 1) & 1) * (GNT * 16 * GKC * 2));
            tdm_load_2d(B + (size_t)tileN * ldb + (k0 + GKC), lds_off,
                        GNT * 16, GKC, ldb);
            __builtin_amdgcn_s_wait_tensorcnt(1);   // chunk ch landed
        } else {
            __builtin_amdgcn_s_wait_tensorcnt(0);   // last chunk landed
        }
        // Launder the LDS pointer + memory clobber: the TDM writes LDS behind
        // the compiler's back (descriptor carries the address as an integer),
        // so force the B-fragment loads to be real and ordered after the wait.
        // Pointer stays in address_space(3) => loads lower to ds_load_b128.
        asm volatile("" : "+v"(bs) : : "memory");
#else
        copy_slab_sync(B + (size_t)tileN * ldb + k0, bsm[ch & 1], ldb, lane);
#endif

        for (int ks = 0; ks < GKC; ks += 32) {
            // B fragments from LDS: 32 contiguous bytes at row n, col ks+half*16
            FragU bq[GNT];
#pragma unroll
            for (int nt = 0; nt < GNT; ++nt) {
                lds_cui* bp = (lds_cui*)
                    (bs + (nt * 16 + l16) * GKC + ks + half * 16);
#pragma unroll
                for (int v = 0; v < 8; ++v) bq[nt].u[v] = bp[v];
            }
            // A fragments direct from global (A is small & cache-hot):
            // two 16B chunks at +half*16 and +half*16+8 elements
            FragU aq[GMT];
#pragma unroll
            for (int mt = 0; mt < GMT; ++mt) {
                const unsigned short* ap =
                    A + (size_t)(tileM + mt * 16 + l16) * lda + k0 + ks;
                const unsigned int* lo = (const unsigned int*)(ap + half * 16);
                const unsigned int* hi = (const unsigned int*)(ap + half * 16 + 8);
#pragma unroll
                for (int v = 0; v < 4; ++v) aq[mt].u[v] = lo[v];
#pragma unroll
                for (int v = 0; v < 4; ++v) aq[mt].u[4 + v] = hi[v];
            }
#pragma unroll
            for (int nt = 0; nt < GNT; ++nt)
#pragma unroll
                for (int mt = 0; mt < GMT; ++mt)
                    acc[mt][nt] = __builtin_amdgcn_wmma_f32_16x16x32_bf16(
                        false, aq[mt].b, false, bq[nt].b,
                        (short)0, acc[mt][nt], false, false);
        }
    }

    // epilogue
#pragma unroll
    for (int nt = 0; nt < GNT; ++nt) {
        const int n = tileN + nt * 16 + l16;
        float bsum = 0.f;
        if (bias0) bsum += bias0[n];
        if (bias1) bsum += bias1[n];
#pragma unroll
        for (int mt = 0; mt < GMT; ++mt) {
#pragma unroll
            for (int v = 0; v < 8; ++v) {
                const int m = tileM + mt * 16 + v + half * 8;
                const long long ci = (long long)m * ldc + n;
                float val = acc[mt][nt][v] + bsum;
                if (accumulate) val += C[ci];
                C[ci] = val;
            }
        }
    }
}

// ---------------------------------------------------------------------------
// Attention step (one block per batch element, 256 threads):
//   score_l = v_a . tanh(a1[b,l,:] + a2[b,:]) + b_v ; softmax over L
//   context[b,:] = sum_l w_l * features[b,l,:]
//   xbuf[b] = [ embed[b,t,:], context[b,:] ]  (bf16)   ; attw out (fp32)
// ---------------------------------------------------------------------------
__global__ void attention_step_kernel(const float* __restrict__ a1,       // [B,L,H]
                                      const float* __restrict__ a2,       // [B,H]
                                      const float* __restrict__ features, // [B,L,F]
                                      const float* __restrict__ v_a,      // [H]
                                      const float* __restrict__ b_v,      // [1]
                                      const unsigned short* __restrict__ embed_bf, // [B,T,E]
                                      int t,
                                      unsigned short* __restrict__ xbuf,  // [B,E+F]
                                      float* __restrict__ attw) {         // [B,T,L]
    const int b    = blockIdx.x;
    const int tid  = threadIdx.x;
    const int lane = tid & 31;
    const int wave = tid >> 5;                       // 8 waves

    __shared__ float s_sh[Ll];

    for (int l = wave; l < Ll; l += 8) {
        const float* a1p = a1 + ((size_t)b * Ll + l) * Hh;
        const float* a2p = a2 + (size_t)b * Hh;
        float acc = 0.f;
        for (int d = lane; d < Hh; d += 32)
            acc += v_a[d] * tanhf(a1p[d] + a2p[d]);
#pragma unroll
        for (int ofs = 16; ofs > 0; ofs >>= 1)
            acc += __shfl_down(acc, ofs, 32);
        if (lane == 0) s_sh[l] = acc + b_v[0];
    }
    __syncthreads();

    if (tid == 0) {
        float mx = s_sh[0];
        for (int l = 1; l < Ll; ++l) mx = fmaxf(mx, s_sh[l]);
        float sum = 0.f;
        for (int l = 0; l < Ll; ++l) { float e = __expf(s_sh[l] - mx); s_sh[l] = e; sum += e; }
        float inv = 1.0f / sum;
        for (int l = 0; l < Ll; ++l) s_sh[l] *= inv;
    }
    __syncthreads();

    if (tid < Ll)
        attw[((size_t)b * Tt + t) * Ll + tid] = s_sh[tid];

    for (int f = tid; f < Ff; f += 256) {
        float c = 0.f;
        const float* fp = features + (size_t)b * Ll * Ff + f;
        for (int l = 0; l < Ll; ++l)
            c += s_sh[l] * fp[(size_t)l * Ff];
        xbuf[(size_t)b * (Ee + Ff) + Ee + f] = f32_to_bf16(c);
    }
    for (int e = tid; e < Ee; e += 256)
        xbuf[(size_t)b * (Ee + Ff) + e] = embed_bf[((size_t)b * Tt + t) * Ee + e];
}

// ------------------------- LSTM pointwise update ----------------------------
__global__ void lstm_pointwise_kernel(const float* __restrict__ gates, // [B,4H]
                                      float* __restrict__ h,
                                      float* __restrict__ c,
                                      unsigned short* __restrict__ h_bf) {
    int i = blockIdx.x * blockDim.x + threadIdx.x;   // over B*H
    if (i >= Bb * Hh) return;
    int j = i % Hh;
    int b = i / Hh;
    const float* g = gates + (size_t)b * 4 * Hh;
    float gi = g[j];
    float gf = g[Hh + j];
    float gg = g[2 * Hh + j];
    float go = g[3 * Hh + j];
    float cn = sigmoidf_(gf) * c[i] + sigmoidf_(gi) * tanhf(gg);
    float hn = sigmoidf_(go) * tanhf(cn);
    c[i] = cn;
    h[i] = hn;
    h_bf[i] = f32_to_bf16(hn);
}

// ---------------------------------------------------------------------------
extern "C" void kernel_launch(void* const* d_in, const int* in_sizes, int n_in,
                              void* d_out, int out_size, void* d_ws, size_t ws_size,
                              hipStream_t stream) {
    (void)in_sizes; (void)n_in; (void)out_size; (void)ws_size;

    const float* features  = (const float*)d_in[0];
    const int*   captions  = (const int*)  d_in[1];
    const float* emb_table = (const float*)d_in[2];
    const float* W_a  = (const float*)d_in[3];
    const float* b_a  = (const float*)d_in[4];
    const float* U_a  = (const float*)d_in[5];
    const float* b_u  = (const float*)d_in[6];
    const float* v_a  = (const float*)d_in[7];
    const float* b_v  = (const float*)d_in[8];
    const float* W_ih = (const float*)d_in[9];
    const float* b_ih = (const float*)d_in[10];
    const float* W_hh = (const float*)d_in[11];
    const float* b_hh = (const float*)d_in[12];
    const float* fc_W = (const float*)d_in[13];
    const float* fc_b = (const float*)d_in[14];
    const float* Wh0  = (const float*)d_in[15];
    const float* bh0  = (const float*)d_in[16];
    const float* Wc0  = (const float*)d_in[17];
    const float* bc0  = (const float*)d_in[18];

    float* outputs = (float*)d_out;                              // [B,T,V]
    float* attw    = (float*)d_out + (size_t)Bb * Tt * Vv;       // [B,T,L]

    // ---- workspace carve-up (256B aligned) --------------------------------
    char* ws = (char*)d_ws;
    size_t off = 0;
    auto alloc = [&](size_t bytes) -> void* {
        void* p = ws + off;
        off += (bytes + 255) & ~(size_t)255;
        return p;
    };
    unsigned short* feat_bf  = (unsigned short*)alloc((size_t)Bb * Ll * Ff * 2);
    unsigned short* W_a_bf   = (unsigned short*)alloc((size_t)Hh * Ff * 2);
    unsigned short* U_a_bf   = (unsigned short*)alloc((size_t)Hh * Hh * 2);
    unsigned short* W_ih_bf  = (unsigned short*)alloc((size_t)4 * Hh * (Ee + Ff) * 2);
    unsigned short* W_hh_bf  = (unsigned short*)alloc((size_t)4 * Hh * Hh * 2);
    unsigned short* fc_W_bf  = (unsigned short*)alloc((size_t)Vv * Hh * 2);
    unsigned short* Wh0_bf   = (unsigned short*)alloc((size_t)Hh * Ff * 2);
    unsigned short* Wc0_bf   = (unsigned short*)alloc((size_t)Hh * Ff * 2);
    unsigned short* mean_bf  = (unsigned short*)alloc((size_t)Bb * Ff * 2);
    unsigned short* embed_bf = (unsigned short*)alloc((size_t)Bb * Tt * Ee * 2);
    unsigned short* xbuf     = (unsigned short*)alloc((size_t)Bb * (Ee + Ff) * 2);
    unsigned short* h_bf     = (unsigned short*)alloc((size_t)Bb * Hh * 2);
    float* h     = (float*)alloc((size_t)Bb * Hh * 4);
    float* c     = (float*)alloc((size_t)Bb * Hh * 4);
    float* a1    = (float*)alloc((size_t)Bb * Ll * Hh * 4);
    float* a2    = (float*)alloc((size_t)Bb * Hh * 4);
    float* gates = (float*)alloc((size_t)Bb * 4 * Hh * 4);

    auto cvt = [&](const float* src, unsigned short* dst, size_t n) {
        cvt_f32_bf16_kernel<<<dim3((unsigned)((n + 255) / 256)), dim3(256), 0, stream>>>(
            src, dst, (int)n);
    };
    auto gemm = [&](const unsigned short* A, int lda, const unsigned short* Bm, int ldb,
                    float* C, long long ldc, const float* bias0, const float* bias1,
                    int M, int N, int K, int accum) {
        gemm_bf16_wmma_kernel<<<dim3((unsigned)(N / (16 * GNT)),
                                     (unsigned)(M / (16 * GMT))),
                                dim3(32), 0, stream>>>(
            A, lda, Bm, ldb, C, ldc, bias0, bias1, K, accum);
    };

    // ---- one-shot setup ---------------------------------------------------
    cvt(features, feat_bf, (size_t)Bb * Ll * Ff);
    cvt(W_a,  W_a_bf,  (size_t)Hh * Ff);
    cvt(U_a,  U_a_bf,  (size_t)Hh * Hh);
    cvt(W_ih, W_ih_bf, (size_t)4 * Hh * (Ee + Ff));
    cvt(W_hh, W_hh_bf, (size_t)4 * Hh * Hh);
    cvt(fc_W, fc_W_bf, (size_t)Vv * Hh);
    cvt(Wh0,  Wh0_bf,  (size_t)Hh * Ff);
    cvt(Wc0,  Wc0_bf,  (size_t)Hh * Ff);

    gather_embed_kernel<<<dim3((Bb * Tt * Ee + 255) / 256), dim3(256), 0, stream>>>(
        emb_table, captions, embed_bf);
    mean_features_kernel<<<dim3((Bb * Ff + 255) / 256), dim3(256), 0, stream>>>(
        features, mean_bf);

    // h0 = mean @ Wh0^T + bh0 ; c0 = mean @ Wc0^T + bc0
    gemm(mean_bf, Ff, Wh0_bf, Ff, h, Hh, bh0, nullptr, Bb, Hh, Ff, 0);
    gemm(mean_bf, Ff, Wc0_bf, Ff, c, Hh, bc0, nullptr, Bb, Hh, Ff, 0);
    // a1 = features @ W_a^T + b_a   (hoisted out of the scan)  [B*L, H]
    gemm(feat_bf, Ff, W_a_bf, Ff, a1, Hh, b_a, nullptr, Bb * Ll, Hh, Ff, 0);
    // bf16 copy of h0 for the first step's GEMMs
    cvt(h, h_bf, (size_t)Bb * Hh);

    // ---- sequential scan over T steps ------------------------------------
    for (int t = 0; t < Tt; ++t) {
        // a2 = h @ U_a^T + b_u            [B,H]
        gemm(h_bf, Hh, U_a_bf, Hh, a2, Hh, b_u, nullptr, Bb, Hh, Hh, 0);
        // attention: scores/softmax/context + build x = [emb_t, context]
        attention_step_kernel<<<dim3(Bb), dim3(256), 0, stream>>>(
            a1, a2, features, v_a, b_v, embed_bf, t, xbuf, attw);
        // gates = x @ W_ih^T + b_ih + b_hh            [B,4H]
        gemm(xbuf, Ee + Ff, W_ih_bf, Ee + Ff, gates, 4 * Hh, b_ih, b_hh,
             Bb, 4 * Hh, Ee + Ff, 0);
        // gates += h @ W_hh^T
        gemm(h_bf, Hh, W_hh_bf, Hh, gates, 4 * Hh, nullptr, nullptr,
             Bb, 4 * Hh, Hh, 1);
        // LSTM pointwise -> h, c (and bf16 h for the following GEMMs)
        lstm_pointwise_kernel<<<dim3((Bb * Hh + 255) / 256), dim3(256), 0, stream>>>(
            gates, h, c, h_bf);
        // out_t = h_new @ fc_W^T + fc_b -> outputs[:, t, :] (row stride T*V)
        gemm(h_bf, Hh, fc_W_bf, Hh, outputs + (size_t)t * Vv, (long long)Tt * Vv,
             fc_b, nullptr, Bb, Vv, Hh, 0);
    }
}
